// GPT2Attention_8478265442339
// MI455X (gfx1250) — matrix operand order repaired and checked by reading, not compiled, not run on hardware
//
#include <hip/hip_runtime.h>
#include <hip/hip_bf16.h>

// ---------------------------------------------------------------------------
// GPT-2 attention for MI455X (gfx1250): all GEMMs via v_wmma_f32_16x16x32_bf16
//   1) qkv = hs @ c_attn_w + b        (f32 -> bf16 on the fly, packed LDS stores)
//   2) flash attention (causal, -10000 mask, online softmax, async K staging)
//   3) out = attn @ c_proj_w + b      (bf16 A staged via global_load_async_to_lds)
// ---------------------------------------------------------------------------

typedef __attribute__((ext_vector_type(16))) __bf16 v16bf;
typedef __attribute__((ext_vector_type(8)))  float  v8f;

static constexpr int Bb = 4, Ss = 2048, Ee = 1024, Hh = 16;
static constexpr int QKV_STRIDE = 3 * Ee; // 3072

__device__ __forceinline__ __bf16 f2bf(float f) { return (__bf16)f; }

__device__ __forceinline__ uint32_t pack2bf(float x, float y) {
  union { __bf16 h[2]; uint32_t u; } p;
  p.h[0] = (__bf16)x; p.h[1] = (__bf16)y;
  return p.u;
}

__device__ __forceinline__ v8f wmma_bf16(v16bf a, v16bf b, v8f c) {
  return __builtin_amdgcn_wmma_f32_16x16x32_bf16(false, a, false, b,
                                                 (short)0, c, false, false);
}

// CDNA5 async memory->LDS copy (ASYNCcnt-tracked), 16 bytes per lane.
// lds_off = raw LDS byte offset (flat shared addr truncated to [31:0]).
__device__ __forceinline__ void async_copy_b128(void* lds_ptr, const void* gptr) {
  const uint32_t lds_off = (uint32_t)(size_t)lds_ptr;
  asm volatile("global_load_async_to_lds_b128 %0, %1, off"
               :: "v"(lds_off), "v"((uint64_t)(size_t)gptr)
               : "memory");
}
__device__ __forceinline__ void async_wait0() {
  asm volatile("s_wait_asynccnt 0" ::: "memory");
}

// A fragment (16x32 bf16) from LDS, row-major, stride in halfs.
// Lane layout: M = lane&15; elems 0..7 -> K = half*8+e ; 8..15 -> K = 16+half*8+e
__device__ __forceinline__ v16bf frag_a_lds(const __bf16* base, int stride, int lane) {
  const int m = lane & 15, h = (lane >> 4) & 1;
  const __bf16* r = base + m * stride;
  v16bf a;
#pragma unroll
  for (int e = 0; e < 8; ++e) a[e] = r[h * 8 + e];
#pragma unroll
  for (int e = 0; e < 8; ++e) a[8 + e] = r[16 + h * 8 + e];
  return a;
}

// B fragment (32x16 bf16, K x N) from LDS stored *transposed* as [N][K].
// Lane layout: N = lane&15; elems 0..7 -> K = half*8+e ; 8..15 -> K = 16+half*8+e
__device__ __forceinline__ v16bf frag_b_ldsT(const __bf16* baseT, int stride, int lane) {
  const int n = lane & 15, h = (lane >> 4) & 1;
  const __bf16* r = baseT + n * stride;
  v16bf b;
#pragma unroll
  for (int e = 0; e < 8; ++e) b[e] = r[h * 8 + e];
#pragma unroll
  for (int e = 0; e < 8; ++e) b[8 + e] = r[16 + h * 8 + e];
  return b;
}

// ---------------------------------------------------------------------------
// Tiled GEMM: C[M,N] = A[M,K] @ W[K,N] + bias.  Block tile 128x128, 8 waves
// (4 along M x 2 along N), each wave owns a 32x64 tile = 2x4 WMMA accumulators
// -> 8 v_wmma per K-step per wave.  AT = float|__bf16, OT = __bf16|float.
// All dims are exact multiples of the tiles (no guards needed here).
// ---------------------------------------------------------------------------
template <typename AT, typename OT>
__global__ __launch_bounds__(256) void wmma_gemm(const AT* __restrict__ A,
                                                 const float* __restrict__ W,
                                                 const float* __restrict__ bias,
                                                 OT* __restrict__ C,
                                                 int M, int N, int K) {
  __shared__ __bf16 As[128][40];  // 128 x 32 (+pad)
  __shared__ __bf16 Bt[128][40];  // N=128 x K=32 (+pad), transposed

  const int tid = threadIdx.x;
  const int lane = tid & 31;
  const int wave = tid >> 5;
  const int m0 = blockIdx.y * 128;
  const int n0 = blockIdx.x * 128;
  const int wm = (wave >> 1) * 32; // 4 waves along M
  const int wn = (wave & 1) * 64;  // 2 waves along N

  v8f acc[2][4];
#pragma unroll
  for (int i = 0; i < 2; ++i)
#pragma unroll
    for (int j = 0; j < 4; ++j)
#pragma unroll
      for (int e = 0; e < 8; ++e) acc[i][j][e] = 0.f;

  for (int k0 = 0; k0 < K; k0 += 32) {
    // ---- stage A tile ----
    if constexpr (sizeof(AT) == 4) {
      // f32 -> bf16 convert, packed ds_store_b64 (4 halfs per store)
      for (int i = tid; i < 128 * 8; i += 256) {
        const int row = i >> 3, c4 = (i & 7) << 2;
        const float4 v = *(const float4*)&A[(size_t)(m0 + row) * K + k0 + c4];
        uint2 p; p.x = pack2bf(v.x, v.y); p.y = pack2bf(v.z, v.w);
        *(uint2*)&As[row][c4] = p;
      }
    } else {
      // already bf16: async global->LDS, 16B per lane, no VGPR round trip
      for (int i = tid; i < 128 * 4; i += 256) {
        const int row = i >> 2, c8 = (i & 3) << 3;
        async_copy_b128(&As[row][c8], &A[(size_t)(m0 + row) * K + k0 + c8]);
      }
    }
    // ---- stage W tile transposed: Bt[n][k] = W[k0+k][n0+n], packed pairs ----
    for (int i = tid; i < 16 * 128; i += 256) {
      const int k2 = i >> 7, nn = i & 127;
      const float w0 = W[(size_t)(k0 + 2 * k2 + 0) * N + n0 + nn];
      const float w1 = W[(size_t)(k0 + 2 * k2 + 1) * N + n0 + nn];
      *(uint32_t*)&Bt[nn][2 * k2] = pack2bf(w0, w1);
    }
    if (k0 + 32 < K) __builtin_prefetch(&W[(size_t)(k0 + 32) * N + n0 + (tid & 127)], 0, 1);
    if constexpr (sizeof(AT) == 2) async_wait0();
    __syncthreads();

    const v16bf a0 = frag_a_lds(&As[wm][0], 40, lane);
    const v16bf a1 = frag_a_lds(&As[wm + 16][0], 40, lane);
#pragma unroll
    for (int j = 0; j < 4; ++j) {
      const v16bf bj = frag_b_ldsT(&Bt[wn + j * 16][0], 40, lane);
      acc[0][j] = wmma_bf16(a0, bj, acc[0][j]);
      acc[1][j] = wmma_bf16(a1, bj, acc[1][j]);
    }
    __syncthreads();
  }

  // ---- epilogue: bias + store (C/D layout: M = r + 8*half, N = lane&15) ----
  const int half = lane >> 4, nl = lane & 15;
#pragma unroll
  for (int ti = 0; ti < 2; ++ti)
#pragma unroll
    for (int tj = 0; tj < 4; ++tj) {
      const int ncol = n0 + wn + tj * 16 + nl;
      const float bv = bias[ncol];
#pragma unroll
      for (int r = 0; r < 8; ++r) {
        const int mrow = m0 + wm + ti * 16 + r + 8 * half;
        const float v = acc[ti][tj][r] + bv;
        if constexpr (sizeof(OT) == 2)
          C[(size_t)mrow * N + ncol] = f2bf(v);
        else
          C[(size_t)mrow * N + ncol] = v;
      }
    }
}

// ---------------------------------------------------------------------------
// Flash attention: grid (S/128, H, B), 8 waves; each wave owns 16 query rows.
// qkv layout [B,S,3E] bf16 with per-head columns h*192 + {0..63 q, 64..127 k,
// 128..191 v}.  KV processed in 32-key blocks; scores via 2 WMMA (hd=64),
// P@V via 4 WMMA (dv=64).  K staged with async global->LDS; V transposed
// manually.  Output merged-heads bf16 [B,S,E].
// ---------------------------------------------------------------------------
__global__ __launch_bounds__(256) void flash_attn(const __bf16* __restrict__ qkv,
                                                  __bf16* __restrict__ out) {
  __shared__ __bf16 Kl[32][72];     // [key][hd]  == B^T layout for Q@K^T
  __shared__ __bf16 Vt[64][40];     // [hd][key]  == B^T layout for P@V
  __shared__ __bf16 Pl[8][16][40];  // per-wave P tile [16 q][32 k]

  const int tid = threadIdx.x, lane = tid & 31, wave = tid >> 5;
  const int qb = blockIdx.x * 128;
  const int h = blockIdx.y, b = blockIdx.z;
  const int qrow0 = qb + wave * 16;
  const int half = lane >> 4, nl = lane & 15;
  const float scale = 0.125f; // 1/sqrt(64)

  const __bf16* hbase = qkv + (size_t)b * Ss * QKV_STRIDE + h * 192;

  // Q fragments (A layout) loaded once: row = qrow0 + (lane&15)
  v16bf aq[2];
#pragma unroll
  for (int c = 0; c < 2; ++c) {
    const __bf16* qr = hbase + (size_t)(qrow0 + nl) * QKV_STRIDE + c * 32;
#pragma unroll
    for (int e = 0; e < 8; ++e) aq[c][e] = qr[half * 8 + e];
#pragma unroll
    for (int e = 0; e < 8; ++e) aq[c][8 + e] = qr[16 + half * 8 + e];
  }

  v8f o[4];
  float mrow[8], lrow[8];
#pragma unroll
  for (int t = 0; t < 4; ++t)
#pragma unroll
    for (int e = 0; e < 8; ++e) o[t][e] = 0.f;
#pragma unroll
  for (int r = 0; r < 8; ++r) { mrow[r] = -1e30f; lrow[r] = 0.f; }

  const int kend = qb + 128; // causal bound for the whole block
  for (int kb = 0; kb < kend; kb += 32) {
    // ---- stage K (async, row-major) and V (transposed): 256 x 8 halfs ----
    {
      const int row = tid >> 3, c8 = (tid & 7) << 3;
      async_copy_b128(&Kl[row][c8],
                      hbase + (size_t)(kb + row) * QKV_STRIDE + 64 + c8);
      const __bf16* vsrc = hbase + (size_t)(kb + row) * QKV_STRIDE + 128 + c8;
      uint4 vv = *(const uint4*)vsrc;
      const __bf16* vh = (const __bf16*)&vv;
#pragma unroll
      for (int j = 0; j < 8; ++j) Vt[c8 + j][row] = vh[j];
    }
    async_wait0();
    __syncthreads();

    if (kb <= qrow0 + 15) { // wave-uniform causal skip
      // ---- scores: two 16x16 tiles, contraction hd=64 (2 WMMA each) ----
      v8f s0, s1;
#pragma unroll
      for (int e = 0; e < 8; ++e) { s0[e] = 0.f; s1[e] = 0.f; }
      s0 = wmma_bf16(aq[0], frag_b_ldsT(&Kl[0][0], 72, lane), s0);
      s0 = wmma_bf16(aq[1], frag_b_ldsT(&Kl[0][32], 72, lane), s0);
      s1 = wmma_bf16(aq[0], frag_b_ldsT(&Kl[16][0], 72, lane), s1);
      s1 = wmma_bf16(aq[1], frag_b_ldsT(&Kl[16][32], 72, lane), s1);

      // ---- scale + causal(-10000) + online softmax per query row ----
#pragma unroll
      for (int r = 0; r < 8; ++r) {
        const int qg = qrow0 + r + 8 * half;
        float v0 = s0[r] * scale; if (kb + nl > qg)      v0 = -10000.f;
        float v1 = s1[r] * scale; if (kb + 16 + nl > qg) v1 = -10000.f;
        float rmax = fmaxf(v0, v1);
#pragma unroll
        for (int m = 1; m < 16; m <<= 1) rmax = fmaxf(rmax, __shfl_xor(rmax, m, 32));
        const float mnew = fmaxf(mrow[r], rmax);
        const float corr = __expf(mrow[r] - mnew);
        const float p0 = __expf(v0 - mnew);
        const float p1 = __expf(v1 - mnew);
        float rsum = p0 + p1;
#pragma unroll
        for (int m = 1; m < 16; m <<= 1) rsum += __shfl_xor(rsum, m, 32);
        lrow[r] = lrow[r] * corr + rsum;
        mrow[r] = mnew;
#pragma unroll
        for (int t = 0; t < 4; ++t) o[t][r] *= corr;
        // C-layout -> LDS so it can be re-read in A-layout
        Pl[wave][r + 8 * half][nl] = f2bf(p0);
        Pl[wave][r + 8 * half][16 + nl] = f2bf(p1);
      }

      // ---- P @ V: contraction over 32 keys (1 WMMA per 16-wide d chunk) ----
      const v16bf ap = frag_a_lds(&Pl[wave][0][0], 40, lane);
#pragma unroll
      for (int t = 0; t < 4; ++t)
        o[t] = wmma_bf16(ap, frag_b_ldsT(&Vt[t * 16][0], 40, lane), o[t]);
    }
    __syncthreads();
  }

  // ---- normalize and store merged-heads bf16 [B,S,E] ----
#pragma unroll
  for (int t = 0; t < 4; ++t)
#pragma unroll
    for (int r = 0; r < 8; ++r) {
      const int row = qrow0 + r + 8 * half;
      const int d = t * 16 + nl;
      out[((size_t)b * Ss + row) * Ee + h * 64 + d] = f2bf(o[t][r] / lrow[r]);
    }
}

// ---------------------------------------------------------------------------
extern "C" void kernel_launch(void* const* d_in, const int* in_sizes, int n_in,
                              void* d_out, int out_size, void* d_ws, size_t ws_size,
                              hipStream_t stream) {
  (void)in_sizes; (void)n_in; (void)out_size; (void)ws_size;
  const float* hs     = (const float*)d_in[0];
  const float* attn_w = (const float*)d_in[1];
  const float* attn_b = (const float*)d_in[2];
  const float* proj_w = (const float*)d_in[3];
  const float* proj_b = (const float*)d_in[4];
  float* out = (float*)d_out;

  __bf16* qkv      = (__bf16*)d_ws;                   // [B,S,3E] bf16 (~48 MB)
  __bf16* attn_out = qkv + (size_t)Bb * Ss * 3 * Ee;  // [B,S,E]  bf16 (~16 MB)

  const int M = Bb * Ss; // 8192

  // 1) QKV projection: [8192,1024] x [1024,3072] -> bf16 qkv
  dim3 g1((3 * Ee) / 128, M / 128);
  wmma_gemm<float, __bf16><<<g1, dim3(256), 0, stream>>>(hs, attn_w, attn_b, qkv,
                                                         M, 3 * Ee, Ee);
  // 2) causal flash attention -> bf16 attn_out (merged heads)
  dim3 g2(Ss / 128, Hh, Bb);
  flash_attn<<<g2, dim3(256), 0, stream>>>(qkv, attn_out);

  // 3) output projection: [8192,1024] x [1024,1024] -> f32 out
  dim3 g3(Ee / 128, M / 128);
  wmma_gemm<__bf16, float><<<g3, dim3(256), 0, stream>>>(attn_out, proj_w, proj_b,
                                                         out, M, Ee, Ee);
}
